// New_Coref_50886772523284
// MI455X (gfx1250) — matrix operand order repaired and checked by reading, not compiled, not run on hardware
//
#include <hip/hip_runtime.h>

typedef unsigned short u16;
typedef __attribute__((ext_vector_type(16))) __bf16 v16bf;
typedef __attribute__((ext_vector_type(8)))  float  v8f;

#define NSPANS  2000
#define DSPAN   1220
#define PAIRS   65536
#define KIN     3720          // 3*1220 + 60
#define KPAD    3744          // 117 * 32
#define KSTEPS  117
#define HID     150
#define NPAD    160
#define NTILES  10
#define KSTEPS2 5             // 160 / 32
#define CHUNK   13            // ksteps staged per LDS chunk; 117 = 9*13
#define NCHUNK  9
#define BLOCK   320           // 10 waves (wave32)

// workspace byte offsets (all 256-aligned)
#define OFF_GBF  0u                         // 2000*1220 u16 = 4,880,000 B
#define OFF_GSQ  4880128u                   // 2000*1220 u16
#define OFF_W1P  9760256u                   // 10*117*512 u16 = 1,198,080 B
#define OFF_W2P  10958336u                  // 10*5*512 u16 = 51,200 B

__device__ __forceinline__ u16 f2bf(float f) {
  unsigned u = __builtin_bit_cast(unsigned, f);
  u += 0x7FFFu + ((u >> 16) & 1u);          // round-to-nearest-even
  return (u16)(u >> 16);
}

__device__ __forceinline__ int dist_bin(int d) {
  // BINS = [1,2,3,4,5,8,16,32,64]; searchsorted(right)-1, clipped
  return (d>=2)+(d>=3)+(d>=4)+(d>=5)+(d>=8)+(d>=16)+(d>=32)+(d>=64);
}

// Slot of element (row m, k%32 == r) inside a 512-entry (1024 B) A-operand
// k-step block.  16-bit A layout: lane<16 holds K {0..7,16..23}, lane>=16
// holds K {8..15,24..31}; 16 bf16 contiguous per lane.
__device__ __forceinline__ int a_slot(int m, int r) {
  int lane, j;
  if      (r <  8) { lane = m;      j = r;      }
  else if (r < 16) { lane = m + 16; j = r - 8;  }
  else if (r < 24) { lane = m;      j = r - 8;  }
  else             { lane = m + 16; j = r - 16; }
  return lane * 16 + j;
}

// ---------------- prologue kernels ----------------

__global__ void k_convert_g(const float* __restrict__ g,
                            u16* __restrict__ gbf, u16* __restrict__ gsq, int n) {
  int i = blockIdx.x * blockDim.x + threadIdx.x;
  if (i < n) {
    float f = g[i];
    gbf[i] = f2bf(f);
    gsq[i] = f2bf(f * f);
  }
}

// Pack W (rowsK x 150, row-major) into WMMA B-operand tiles:
// out[((nt*ksteps + ks)*32 + lane)*16 + j] = W[kk*150 + n],
// n = nt*16 + (lane&15), kk = ks*32 + (lane>>4)*8 + (j&7) + (j>>3)*16   (zero-padded).
__global__ void k_pack_w(const float* __restrict__ W, u16* __restrict__ out,
                         int ksteps, int rowsK, int total) {
  int idx = blockIdx.x * blockDim.x + threadIdx.x;
  if (idx >= total) return;
  int j    = idx & 15;
  int lane = (idx >> 4) & 31;
  int rest = idx >> 9;
  int ks   = rest % ksteps;
  int nt   = rest / ksteps;
  int n    = nt * 16 + (lane & 15);
  int kk   = ks * 32 + (lane >> 4) * 8 + (j & 7) + (j >> 3) * 16;
  float v  = (n < HID && kk < rowsK) ? W[kk * HID + n] : 0.0f;
  out[idx] = f2bf(v);
}

// ---------------- main fused kernel ----------------

__global__ __launch_bounds__(BLOCK)
void k_coref_main(const u16* __restrict__ gbf, const u16* __restrict__ gsq,
                  const u16* __restrict__ w1p, const u16* __restrict__ w2p,
                  const int* __restrict__ mention, const int* __restrict__ antecedent,
                  const int* __restrict__ dists, const int* __restrict__ genres,
                  const int* __restrict__ speakers,
                  const float* __restrict__ dist_emb, const float* __restrict__ genre_emb,
                  const float* __restrict__ speaker_emb,
                  const float* __restrict__ b1, const float* __restrict__ b2,
                  const float* __restrict__ W3, const float* __restrict__ b3,
                  const float* __restrict__ s_m, float* __restrict__ out) {
  __shared__ __align__(32) u16 sA[CHUNK * 512];     // 13,312 B staged A chunk
  __shared__ __align__(32) u16 sH1[KSTEPS2 * 512];  //  5,120 B layer-1 activations
  __shared__ float sS[16];
  __shared__ int   sMid[16], sAid[16];

  const int tid  = threadIdx.x;
  const int lane = tid & 31;
  const int nt   = tid >> 5;                 // wave id == HID column tile
  const int row0 = blockIdx.x * 16;

  if (tid < 16) {
    sMid[tid] = mention[row0 + tid];
    sAid[tid] = antecedent[row0 + tid];
    sS[tid]   = 0.0f;
  }

  v8f c1 = {};

  for (int c = 0; c < NCHUNK; ++c) {
    const int kb = c * (CHUNK * 32);         // global K base of this chunk
    __syncthreads();                          // prev chunk consumed / ids ready

    // ---- cooperative gather+convert of A chunk into LDS (A-operand layout)
    for (int e = tid; e < 16 * CHUNK * 32; e += BLOCK) {
      int m    = e / (CHUNK * 32);
      int kloc = e - m * (CHUNK * 32);
      int k    = kb + kloc;
      u16 val;
      if (k < DSPAN) {
        val = gbf[sMid[m] * DSPAN + k];
      } else if (k < 2 * DSPAN) {
        val = gbf[sAid[m] * DSPAN + (k - DSPAN)];
      } else if (k < 3 * DSPAN) {
        val = gsq[sAid[m] * DSPAN + (k - 2 * DSPAN)];
      } else if (k < KIN) {
        int cphi = k - 3 * DSPAN;
        int p    = row0 + m;
        float f;
        if (cphi < 20)      f = dist_emb[dist_bin(dists[p]) * 20 + cphi];
        else if (cphi < 40) f = genre_emb[genres[p] * 20 + (cphi - 20)];
        else                f = speaker_emb[speakers[p] * 20 + (cphi - 40)];
        val = f2bf(f);
      } else {
        val = 0;                              // K padding 3720..3743
      }
      sA[(kloc >> 5) * 512 + a_slot(m, kloc & 31)] = val;
    }
    __syncthreads();

    // ---- layer-1 WMMA over this chunk (all waves, EXEC all-ones)
    for (int ksl = 0; ksl < CHUNK; ++ksl) {
      v16bf a = *(const v16bf*)&sA[ksl * 512 + lane * 16];
      v16bf b = *(const v16bf*)(w1p + ((nt * KSTEPS + (c * CHUNK + ksl)) * 32 + lane) * 16);
      c1 = __builtin_amdgcn_wmma_f32_16x16x32_bf16(false, a, false, b,
                                                   (short)0, c1, false, false);
    }
  }

  // ---- bias + ReLU, repack h1 as A operand for layer 2
  const int nglob = nt * 16 + (lane & 15);   // output column this lane holds
  const int mbase = (lane >> 4) * 8;
  const float bias1 = (nglob < HID) ? b1[nglob] : 0.0f;
#pragma unroll
  for (int i = 0; i < 8; ++i) {
    float h = fmaxf(c1[i] + bias1, 0.0f);
    sH1[(nglob >> 5) * 512 + a_slot(mbase + i, nglob & 31)] = f2bf(h);
  }
  __syncthreads();

  // ---- layer-2 WMMA (K = 160)
  v8f c2 = {};
#pragma unroll
  for (int ks = 0; ks < KSTEPS2; ++ks) {
    v16bf a = *(const v16bf*)&sH1[ks * 512 + lane * 16];
    v16bf b = *(const v16bf*)(w2p + ((nt * KSTEPS2 + ks) * 32 + lane) * 16);
    c2 = __builtin_amdgcn_wmma_f32_16x16x32_bf16(false, a, false, b,
                                                 (short)0, c2, false, false);
  }

  // ---- layer-3: s_ij[m] = sum_n relu(h2)[m][n] * W3[n]
  const float bias2 = (nglob < HID) ? b2[nglob] : 0.0f;
  const float w3v   = (nglob < HID) ? W3[nglob] : 0.0f;
#pragma unroll
  for (int i = 0; i < 8; ++i) {
    float h = fmaxf(c2[i] + bias2, 0.0f);
    atomicAdd(&sS[mbase + i], h * w3v);
  }
  __syncthreads();

  if (tid < 16) {
    int p = row0 + tid;
    out[p] = sS[tid] + b3[0] + s_m[sMid[tid]] + s_m[sAid[tid]];
  }
}

// ---------------- host launcher ----------------

extern "C" void kernel_launch(void* const* d_in, const int* in_sizes, int n_in,
                              void* d_out, int out_size, void* d_ws, size_t ws_size,
                              hipStream_t stream) {
  const float* g_i         = (const float*)d_in[0];
  const float* s_m         = (const float*)d_in[1];
  const int*   mention     = (const int*)d_in[2];
  const int*   antecedent  = (const int*)d_in[3];
  const int*   dists       = (const int*)d_in[4];
  const int*   genres      = (const int*)d_in[5];
  const int*   speakers    = (const int*)d_in[6];
  const float* dist_emb    = (const float*)d_in[7];
  const float* genre_emb   = (const float*)d_in[8];
  const float* speaker_emb = (const float*)d_in[9];
  const float* W1          = (const float*)d_in[10];
  const float* b1          = (const float*)d_in[11];
  const float* W2          = (const float*)d_in[12];
  const float* b2          = (const float*)d_in[13];
  const float* W3          = (const float*)d_in[14];
  const float* b3          = (const float*)d_in[15];
  float*       out         = (float*)d_out;

  char* ws  = (char*)d_ws;
  u16*  gbf = (u16*)(ws + OFF_GBF);
  u16*  gsq = (u16*)(ws + OFF_GSQ);
  u16*  w1p = (u16*)(ws + OFF_W1P);
  u16*  w2p = (u16*)(ws + OFF_W2P);

  const int ng = NSPANS * DSPAN;             // 2,440,000
  k_convert_g<<<(ng + 255) / 256, 256, 0, stream>>>(g_i, gbf, gsq, ng);

  const int w1total = NTILES * KSTEPS * 512; // 599,040
  k_pack_w<<<(w1total + 255) / 256, 256, 0, stream>>>(W1, w1p, KSTEPS, KIN, w1total);

  const int w2total = NTILES * KSTEPS2 * 512; // 25,600
  k_pack_w<<<(w2total + 255) / 256, 256, 0, stream>>>(W2, w2p, KSTEPS2, HID, w2total);

  k_coref_main<<<PAIRS / 16, BLOCK, 0, stream>>>(
      gbf, gsq, w1p, w2p, mention, antecedent, dists, genres, speakers,
      dist_emb, genre_emb, speaker_emb, b1, b2, W3, b3, s_m, out);
}
